// GINOutlierGNN_2035814498861
// MI455X (gfx1250) — matrix operand order repaired and checked by reading, not compiled, not run on hardware
//
#include <hip/hip_runtime.h>
#include <hip/hip_bf16.h>

typedef _Float16 h16;
typedef _Float16 v16h __attribute__((ext_vector_type(16)));
typedef _Float16 v8h  __attribute__((ext_vector_type(8)));
typedef float    v8f  __attribute__((ext_vector_type(8)));

#define NN   50000
#define EE   600000
#define DIN  128
#define HH   256
#define GG   128

// ---------------------------------------------------------------- utilities

__global__ void fill_f32(float* __restrict__ p, float v, int n) {
    int i = blockIdx.x * blockDim.x + threadIdx.x;
    if (i < n) p[i] = v;
}

// W: [K, M] f32 row-major  ->  Wt: [M, K] f16 row-major (per-column contiguous K)
__global__ void transpose_f32_to_f16(const float* __restrict__ W,
                                     h16* __restrict__ Wt, int K, int M) {
    int i = blockIdx.x * blockDim.x + threadIdx.x;
    if (i >= K * M) return;
    int m = i / K, k = i % K;
    Wt[i] = (h16)W[(size_t)k * M + m];
}

// z[i] = (f16)(a[i] + b[i])
__global__ void add_cast(const float* __restrict__ a, const float* __restrict__ b,
                         h16* __restrict__ z, int n) {
    int i = blockIdx.x * blockDim.x + threadIdx.x;
    if (i < n) z[i] = (h16)(a[i] + b[i]);
}

// agg[dst[e]][:] += h[src[e]][:]   (float4 chunks, lanes contiguous in features)
__global__ void scatter_add(const float* __restrict__ h, const int* __restrict__ src,
                            const int* __restrict__ dst, float* __restrict__ agg,
                            int E, int K) {
    int chunks = K >> 2;
    int i = blockIdx.x * blockDim.x + threadIdx.x;
    if (i >= E * chunks) return;
    int e = i / chunks;
    int c = (i - e * chunks) << 2;
    int s = src[e], d = dst[e];
    const float4 v = *(const float4*)(h + (size_t)s * K + c);
    float* o = agg + (size_t)d * K + c;
    atomicAdd(o + 0, v.x); atomicAdd(o + 1, v.y);
    atomicAdd(o + 2, v.z); atomicAdd(o + 3, v.w);
}

// ---------------------------------------------------------------- WMMA GEMM
// out = relu(A @ W + bias),  A:[M,K] f16 row-major, Bt = W^T : [HOUT,K] f16.
// One wave computes a 16x64 output tile: 4 f32 accumulators share one A
// fragment per K-step (4x fewer A reads than 16x16 tiling).
//
// A fragment (16-bit 16x32): lane m = lane&15; lane half selects K 0-15 vs
// 16-31; element j -> K = 16*half + j: two contiguous
// 8-half loads. B fragment (32x16): lane n = lane&15, half selects K 0-15 vs
// 16-31; element j -> K = 16*half + j: one contiguous 16-half load from Bt.
// C/D f32 16x16: element r -> M = r + 8*half, N = lane&15.
__global__ void __launch_bounds__(32)
wmma_gemm_relu(const h16* __restrict__ A, const h16* __restrict__ Bt,
               const float* __restrict__ bias,
               h16* __restrict__ outH, float* __restrict__ outF,
               int K, int HOUT) {
    const int lane = threadIdx.x;
    const int half = lane >> 4;
    const int lid  = lane & 15;
    const int row0 = blockIdx.x * 16;
    const int col0 = blockIdx.y * 64;
    const int m = row0 + lid;
    const size_t arow = (size_t)m * K;
    const int kbA = half * 8;
    const int kbB = half * 16;

    const h16* brow[4];
#pragma unroll
    for (int t = 0; t < 4; ++t)
        brow[t] = Bt + (size_t)(col0 + t * 16 + lid) * K + kbB;

    v8f c0 = {}, c1 = {}, c2 = {}, c3 = {};
    for (int kk = 0; kk < K; kk += 32) {
        v8h a_lo = *(const v8h*)(A + arow + kk + kbB);
        v8h a_hi = *(const v8h*)(A + arow + kk + kbB + 8);
        v16h a;
#pragma unroll
        for (int j = 0; j < 8; ++j) { a[j] = a_lo[j]; a[j + 8] = a_hi[j]; }
        v16h b0 = *(const v16h*)(brow[0] + kk);
        v16h b1 = *(const v16h*)(brow[1] + kk);
        v16h b2 = *(const v16h*)(brow[2] + kk);
        v16h b3 = *(const v16h*)(brow[3] + kk);
        c0 = __builtin_amdgcn_wmma_f32_16x16x32_f16(false, a, false, b0, (short)0, c0, false, false);
        c1 = __builtin_amdgcn_wmma_f32_16x16x32_f16(false, a, false, b1, (short)0, c1, false, false);
        c2 = __builtin_amdgcn_wmma_f32_16x16x32_f16(false, a, false, b2, (short)0, c2, false, false);
        c3 = __builtin_amdgcn_wmma_f32_16x16x32_f16(false, a, false, b3, (short)0, c3, false, false);
    }

    v8f acc[4] = {c0, c1, c2, c3};
#pragma unroll
    for (int t = 0; t < 4; ++t) {
        const int n = col0 + t * 16 + lid;
        const float bv = bias[n];
#pragma unroll
        for (int r = 0; r < 8; ++r) {
            int row = row0 + r + half * 8;
            float v = fmaxf(acc[t][r] + bv, 0.0f);
            if (outF) outF[(size_t)row * HOUT + n] = v;
            if (outH) outH[(size_t)row * HOUT + n] = (h16)v;
        }
    }
}

// ---------------------------------------------------------------- pooling

__global__ void count_nodes(const int* __restrict__ batch, float* __restrict__ cnt, int n) {
    int i = blockIdx.x * blockDim.x + threadIdx.x;
    if (i < n) atomicAdd(&cnt[batch[i]], 1.0f);
}

// h >= 0 (post-ReLU) so int-punned atomicMax against 0-init is exact.
__global__ void pool_acc(const float* __restrict__ h, const int* __restrict__ batch,
                         float* __restrict__ sums, float* __restrict__ maxs, int n) {
    const int chunks = HH >> 2;
    int i = blockIdx.x * blockDim.x + threadIdx.x;
    if (i >= n * chunks) return;
    int node = i / chunks;
    int c = (i - node * chunks) << 2;
    int g = batch[node];
    const float4 v = *(const float4*)(h + (size_t)node * HH + c);
    float* sp = sums + (size_t)g * HH + c;
    int*   mp = (int*)(maxs + (size_t)g * HH + c);
    atomicAdd(sp + 0, v.x); atomicAdd(sp + 1, v.y);
    atomicAdd(sp + 2, v.z); atomicAdd(sp + 3, v.w);
    atomicMax(mp + 0, __float_as_int(v.x)); atomicMax(mp + 1, __float_as_int(v.y));
    atomicMax(mp + 2, __float_as_int(v.z)); atomicMax(mp + 3, __float_as_int(v.w));
}

__global__ void build_p(const float* __restrict__ sums, const float* __restrict__ maxs,
                        const float* __restrict__ cnt, h16* __restrict__ p) {
    int i = blockIdx.x * blockDim.x + threadIdx.x;
    if (i >= GG * 2 * HH) return;
    int g = i / (2 * HH), f = i - g * 2 * HH;
    float v = (f < HH) ? sums[(size_t)g * HH + f] / fmaxf(cnt[g], 1.0f)
                       : maxs[(size_t)g * HH + (f - HH)];
    p[i] = (h16)v;
}

// out[g] = dot(q[g,:], Wf2) + bf2   (wave32 shuffle reduction)
__global__ void __launch_bounds__(32)
head_out(const h16* __restrict__ q, const float* __restrict__ Wf2,
         const float* __restrict__ bf2, float* __restrict__ out) {
    int g = blockIdx.x, lane = threadIdx.x;
    float acc = 0.0f;
    for (int h = lane; h < HH; h += 32)
        acc += (float)q[(size_t)g * HH + h] * Wf2[h];
#pragma unroll
    for (int off = 16; off > 0; off >>= 1)
        acc += __shfl_xor(acc, off, 32);
    if (lane == 0) out[g] = acc + bf2[0];
}

// ---------------------------------------------------------------- launch

extern "C" void kernel_launch(void* const* d_in, const int* in_sizes, int n_in,
                              void* d_out, int out_size, void* d_ws, size_t ws_size,
                              hipStream_t stream) {
    const float* x    = (const float*)d_in[0];
    const int*   ei   = (const int*)d_in[1];
    const int*   bat  = (const int*)d_in[2];
    const float* W1   = (const float*)d_in[3];  const float* b1 = (const float*)d_in[4];
    const float* W2   = (const float*)d_in[5];  const float* b2 = (const float*)d_in[6];
    const float* W3   = (const float*)d_in[7];  const float* b3 = (const float*)d_in[8];
    const float* W4   = (const float*)d_in[9];  const float* b4 = (const float*)d_in[10];
    const float* Wf1  = (const float*)d_in[11]; const float* bf1 = (const float*)d_in[12];
    const float* Wf2  = (const float*)d_in[13]; const float* bf2 = (const float*)d_in[14];
    const int* src = ei;
    const int* dst = ei + EE;
    float* out = (float*)d_out;

    char* wsb = (char*)d_ws;
    size_t off = 0;
    auto alloc = [&](size_t bytes) -> void* {
        void* p = wsb + off;
        off = (off + bytes + 255) & ~(size_t)255;
        return p;
    };

    h16*   wt1  = (h16*)  alloc((size_t)HH * DIN * 2);      // [256,128]
    h16*   wt2  = (h16*)  alloc((size_t)HH * HH * 2);       // [256,256]
    h16*   wt3  = (h16*)  alloc((size_t)HH * HH * 2);
    h16*   wt4  = (h16*)  alloc((size_t)HH * HH * 2);
    h16*   wtf1 = (h16*)  alloc((size_t)HH * 2 * HH * 2);   // [256,512]
    float* agg  = (float*)alloc((size_t)NN * HH * 4);
    h16*   zh   = (h16*)  alloc((size_t)NN * HH * 2);
    h16*   yh   = (h16*)  alloc((size_t)NN * HH * 2);
    float* hb   = (float*)alloc((size_t)NN * HH * 4);
    float* sums = (float*)alloc((size_t)GG * HH * 4);
    float* maxs = (float*)alloc((size_t)GG * HH * 4);
    float* cnt  = (float*)alloc((size_t)GG * 4);
    h16*   ph   = (h16*)  alloc((size_t)GG * 2 * HH * 2);
    h16*   qh   = (h16*)  alloc((size_t)GG * HH * 2);

    const int T = 256;
    auto blks = [](int n, int t) { return (n + t - 1) / t; };

    // --- one-time weight transpose to f16 [HOUT, K]
    transpose_f32_to_f16<<<blks(DIN * HH, T), T, 0, stream>>>(W1, wt1, DIN, HH);
    transpose_f32_to_f16<<<blks(HH * HH, T), T, 0, stream>>>(W2, wt2, HH, HH);
    transpose_f32_to_f16<<<blks(HH * HH, T), T, 0, stream>>>(W3, wt3, HH, HH);
    transpose_f32_to_f16<<<blks(HH * HH, T), T, 0, stream>>>(W4, wt4, HH, HH);
    transpose_f32_to_f16<<<blks(2 * HH * HH, T), T, 0, stream>>>(Wf1, wtf1, 2 * HH, HH);

    // --- conv1: agg = scatter(x); z = x+agg; h1 = relu(relu(z@W1+b1)@W2+b2)
    fill_f32<<<blks(NN * DIN, T), T, 0, stream>>>(agg, 0.0f, NN * DIN);
    scatter_add<<<blks(EE * (DIN / 4), T), T, 0, stream>>>(x, src, dst, agg, EE, DIN);
    add_cast<<<blks(NN * DIN, T), T, 0, stream>>>(x, agg, zh, NN * DIN);
    wmma_gemm_relu<<<dim3(NN / 16, HH / 64), 32, 0, stream>>>(zh, wt1, b1, yh, nullptr, DIN, HH);
    wmma_gemm_relu<<<dim3(NN / 16, HH / 64), 32, 0, stream>>>(yh, wt2, b2, nullptr, hb, HH, HH);

    // --- conv2: agg = scatter(h1); z = h1+agg; h2 = relu(relu(z@W3+b3)@W4+b4)
    fill_f32<<<blks(NN * HH, T), T, 0, stream>>>(agg, 0.0f, NN * HH);
    scatter_add<<<blks(EE * (HH / 4), T), T, 0, stream>>>(hb, src, dst, agg, EE, HH);
    add_cast<<<blks(NN * HH, T), T, 0, stream>>>(hb, agg, zh, NN * HH);
    wmma_gemm_relu<<<dim3(NN / 16, HH / 64), 32, 0, stream>>>(zh, wt3, b3, yh, nullptr, HH, HH);
    wmma_gemm_relu<<<dim3(NN / 16, HH / 64), 32, 0, stream>>>(yh, wt4, b4, nullptr, hb, HH, HH);

    // --- pooling (mean + max per graph; h2 >= 0 so 0-init max is exact)
    fill_f32<<<blks(GG * HH, T), T, 0, stream>>>(sums, 0.0f, GG * HH);
    fill_f32<<<blks(GG * HH, T), T, 0, stream>>>(maxs, 0.0f, GG * HH);
    fill_f32<<<1, T, 0, stream>>>(cnt, 0.0f, GG);
    count_nodes<<<blks(NN, T), T, 0, stream>>>(bat, cnt, NN);
    pool_acc<<<blks(NN * (HH / 4), T), T, 0, stream>>>(hb, bat, sums, maxs, NN);
    build_p<<<blks(GG * 2 * HH, T), T, 0, stream>>>(sums, maxs, cnt, ph);

    // --- head: q = relu(p@Wf1+bf1); out = q@Wf2+bf2
    wmma_gemm_relu<<<dim3(GG / 16, HH / 64), 32, 0, stream>>>(ph, wtf1, bf1, qh, nullptr, 2 * HH, HH);
    head_out<<<GG, 32, 0, stream>>>(qh, Wf2, bf2, out);

    (void)in_sizes; (void)n_in; (void)out_size; (void)ws_size;
}
